// TokenEncoderLayer_68384469287336
// MI455X (gfx1250) — matrix operand order repaired and checked by reading, not compiled, not run on hardware
//
#include <hip/hip_runtime.h>

#define DM   512
#define DSn  256
#define DI   512
#define DTR  32
#define DFF  1024
#define BB   8
#define LL   2048
#define MROWS (BB*LL)            // 16384
#define XPN  (DTR + 2*DSn)       // 544

typedef unsigned short u16;
typedef __attribute__((ext_vector_type(16))) __bf16 v16bf;
typedef __attribute__((ext_vector_type(8)))  float  v8f;
typedef int v4i __attribute__((vector_size(16)));   // matches async builtin param type

union BF16x16 { uint4 u[2]; v16bf v; };

// ---- gfx1250 async global->LDS path (probe-confirmed on this toolchain) ----
#define ASYNC_OK (__has_builtin(__builtin_amdgcn_global_load_async_to_lds_b128) && \
                  __has_builtin(__builtin_amdgcn_global_load_async_to_lds_b32)  && \
                  __has_builtin(__builtin_amdgcn_s_wait_asynccnt))
#if ASYNC_OK
#define ASYNC_CP128(dst, src) __builtin_amdgcn_global_load_async_to_lds_b128( \
    (__attribute__((address_space(1))) v4i*)(src),                            \
    (__attribute__((address_space(3))) v4i*)(dst), 0, 0)
#define ASYNC_CP32(dst, src)  __builtin_amdgcn_global_load_async_to_lds_b32(  \
    (__attribute__((address_space(1))) int*)(src),                            \
    (__attribute__((address_space(3))) int*)(dst), 0, 0)
#define ASYNC_WAIT()          __builtin_amdgcn_s_wait_asynccnt(0)
#else
#define ASYNC_CP128(dst, src) (*(uint4*)(dst) = *(const uint4*)(src))
#define ASYNC_CP32(dst, src)  (*(float*)(dst) = *(const float*)(src))
#define ASYNC_WAIT()          ((void)0)
#endif

__device__ __forceinline__ u16 f2bf(float f) {
  union { float f; unsigned int u; } x; x.f = f;
  unsigned int u = x.u;
  return (u16)((u + 0x7FFFu + ((u >> 16) & 1u)) >> 16);
}

// ---------------- elementwise converts ----------------
__global__ void k_f32_to_bf16(const float* __restrict__ s, u16* __restrict__ d, int n) {
  int i = blockIdx.x * 256 + threadIdx.x;
  if (i < n) d[i] = f2bf(s[i]);
}

__global__ void k_f32_to_bf16_strided(const float* __restrict__ s, u16* __restrict__ d,
                                      int rows, int srcStride, int width) {
  int i = blockIdx.x * 256 + threadIdx.x;
  if (i < rows * width) {
    int r = i / width, c = i - r * width;
    d[i] = f2bf(s[(size_t)r * srcStride + c]);
  }
}

// ---------------- WMMA GEMM: C[M,N] = act(A[M,K] @ W[N,K]^T + bias) ----------------
// 256 threads = 8 waves; block tile 128(M) x 64(N). B tile staged in LDS
// (double-buffered via async global->LDS; steady-state loop issues the next
// tile unconditionally, last iteration peeled). A fragments stream from global
// (no cross-wave reuse; contiguous k order k = 16h + s, same as B).
// ACT: 0=none, 1=relu, 2=softplus.
#define BROW 40   // LDS row pitch in halves (80B) to spread banks
template<int ACT>
__global__ __launch_bounds__(256)
void k_gemm_bf16(const u16* __restrict__ A, const u16* __restrict__ W,
                 const float* __restrict__ bias, float* __restrict__ C,
                 int M, int N, int K) {
  __shared__ u16 sB[2][64 * BROW];
  const int tid  = threadIdx.x;
  const int lane = tid & 31;
  const int wave = tid >> 5;
  const int half = lane >> 4;       // 0: lanes 0-15, 1: lanes 16-31
  const int l16  = lane & 15;
  const int m0 = blockIdx.y * 128 + wave * 16;
  const int n0 = blockIdx.x * 64;

  v8f acc[4] = {};

  int rowA = m0 + l16; if (rowA >= M) rowA = M - 1;
  const u16* Arow = A + (size_t)rowA * K;

  // this thread's 16B chunk of the 64x32 B tile: 256 threads x 8 halves = 2048 halves
  const int brow = tid >> 2;            // 0..63 (n within tile)
  const int bk   = (tid & 3) * 8;       // 0,8,16,24 (k within tile, halves)
  int gr = n0 + brow; if (gr >= N) gr = N - 1;
  const u16* Bsrc = W + (size_t)gr * K + bk;
  u16* Bd[2] = { &sB[0][brow * BROW + bk], &sB[1][brow * BROW + bk] };

  auto step = [&](int k0, int bf) {
    // A fragment: lane M=l16, K = half*16+{0..15}
    BF16x16 a;
    a.u[0] = *(const uint4*)(Arow + k0 + half * 16);
    a.u[1] = *(const uint4*)(Arow + k0 + half * 16 + 8);
    // B fragments from LDS (contiguous k order, same as A): lane N=l16, K = half*16+{0..15}
    const u16* bb = sB[bf];
    BF16x16 b0, b1, b2, b3;
    b0.u[0] = *(const uint4*)(bb + ( 0 + l16) * BROW + half * 16);
    b0.u[1] = *(const uint4*)(bb + ( 0 + l16) * BROW + half * 16 + 8);
    b1.u[0] = *(const uint4*)(bb + (16 + l16) * BROW + half * 16);
    b1.u[1] = *(const uint4*)(bb + (16 + l16) * BROW + half * 16 + 8);
    b2.u[0] = *(const uint4*)(bb + (32 + l16) * BROW + half * 16);
    b2.u[1] = *(const uint4*)(bb + (32 + l16) * BROW + half * 16 + 8);
    b3.u[0] = *(const uint4*)(bb + (48 + l16) * BROW + half * 16);
    b3.u[1] = *(const uint4*)(bb + (48 + l16) * BROW + half * 16 + 8);
    acc[0] = __builtin_amdgcn_wmma_f32_16x16x32_bf16(false, a.v, false, b0.v, (short)0, acc[0], false, false);
    acc[1] = __builtin_amdgcn_wmma_f32_16x16x32_bf16(false, a.v, false, b1.v, (short)0, acc[1], false, false);
    acc[2] = __builtin_amdgcn_wmma_f32_16x16x32_bf16(false, a.v, false, b2.v, (short)0, acc[2], false, false);
    acc[3] = __builtin_amdgcn_wmma_f32_16x16x32_bf16(false, a.v, false, b3.v, (short)0, acc[3], false, false);
  };

  // preload tile k0=0 into buffer 0
  ASYNC_CP128(Bd[0], Bsrc);

  int buf = 0;
  int k0 = 0;
  for (; k0 < K - 32; k0 += 32) {       // steady state: unconditional next-tile issue
    ASYNC_WAIT();
    __syncthreads();                    // buf tile resident & published
    ASYNC_CP128(Bd[buf ^ 1], Bsrc + k0 + 32);
    __builtin_prefetch(Arow + k0 + 32, 0, 1);   // global_prefetch_b8 for A
    step(k0, buf);
    buf ^= 1;
  }
  ASYNC_WAIT();                         // peeled last iteration
  __syncthreads();
  step(k0, buf);

#pragma unroll
  for (int nt = 0; nt < 4; nt++) {
    int col = n0 + nt * 16 + l16;
    if (col >= N) continue;
    float bv = bias ? bias[col] : 0.f;
#pragma unroll
    for (int r = 0; r < 8; r++) {
      int row = m0 + half * 8 + r;      // ISA C layout: lanes 16-31 hold M=8..15
      if (row >= M) continue;
      float v = acc[nt][r] + bv;
      if (ACT == 1) v = v > 0.f ? v : 0.f;
      if (ACT == 2) v = (v > 20.f) ? v : log1pf(__expf(v));
      C[(size_t)row * N + col] = v;
    }
  }
}

// ---------------- depthwise causal conv(width 2) + SiLU ----------------
__global__ void k_conv_silu(const float* __restrict__ xz, const float* __restrict__ cw,
                            const float* __restrict__ cb, float* __restrict__ xs,
                            u16* __restrict__ xsbf) {
  size_t idx = (size_t)blockIdx.x * 256 + threadIdx.x;
  if (idx >= (size_t)MROWS * DI) return;
  int d = (int)(idx & (DI - 1));
  size_t bl = idx >> 9;                 // b*L + l
  int l = (int)(bl & (LL - 1));
  float cur  = xz[bl * (2 * DI) + d];
  float prev = (l > 0) ? xz[(bl - 1) * (2 * DI) + d] : 0.f;
  float v = cw[d * 2 + 0] * prev + cw[d * 2 + 1] * cur + cb[d];
  float sv = v / (1.f + __expf(-v));    // silu
  xs[idx] = sv;
  xsbf[idx] = f2bf(sv);
}

// ---------------- selective scan: one wave32 per (b,d) row ----------------
// B_t/C_t staged in LDS (double-buffered, async); steady-state loop prefetches
// step t+1 unconditionally, last step peeled.
__global__ __launch_bounds__(256)
void k_scan(const float* __restrict__ dt, const float* __restrict__ dbl,
            const float* __restrict__ xs, const float* __restrict__ A_log,
            float* __restrict__ y) {
  __shared__ float sB[2][DSn], sC[2][DSn];
  int tid = threadIdx.x;
  int lane = tid & 31, wave = tid >> 5;
  int b = blockIdx.x >> 6;
  int d = ((blockIdx.x & 63) << 3) + wave;   // 8 d-rows per block
  int n0 = lane * 8;                         // 8 states per lane -> 256 per wave
  float a2[8], h[8];
#pragma unroll
  for (int i = 0; i < 8; i++) {
    a2[i] = -__expf(A_log[(size_t)d * DSn + n0 + i]) * 1.44269504088896340736f; // A*log2(e)
    h[i] = 0.f;
  }
  const size_t base_bl = (size_t)b * LL;

  auto stepc = [&](int t, int bf) {
    size_t row = base_bl + t;
    float dt_t = dt[row * DI + d];
    float x_t  = xs[row * DI + d];
    float dtx  = dt_t * x_t;
    float part = 0.f;
#pragma unroll
    for (int i = 0; i < 8; i++) {
      h[i] = exp2f(dt_t * a2[i]) * h[i] + dtx * sB[bf][n0 + i];
      part += h[i] * sC[bf][n0 + i];
    }
#pragma unroll
    for (int o = 16; o > 0; o >>= 1) part += __shfl_xor(part, o, 32);
    if (lane == 0) y[row * DI + d] = part;
  };

  { // preload t=0 into buffer 0
    const float* p = dbl + base_bl * XPN + DTR;
    ASYNC_CP32(&sB[0][tid], p + tid);
    ASYNC_CP32(&sC[0][tid], p + DSn + tid);
  }
  int buf = 0;
  for (int t = 0; t < LL - 1; t++) {     // steady state: unconditional next-step issue
    ASYNC_WAIT();
    __syncthreads();                     // buffer `buf` resident & published
    const float* p = dbl + (base_bl + t + 1) * XPN + DTR;
    ASYNC_CP32(&sB[buf ^ 1][tid], p + tid);
    ASYNC_CP32(&sC[buf ^ 1][tid], p + DSn + tid);
    stepc(t, buf);
    buf ^= 1;
  }
  ASYNC_WAIT();                          // peeled last step
  __syncthreads();
  stepc(LL - 1, buf);
}

// ---------------- y = (y + xs*D) * silu(z), emit bf16 for out_proj ----------------
__global__ void k_gate(const float* __restrict__ y, const float* __restrict__ xs,
                       const float* __restrict__ xz, const float* __restrict__ Dsk,
                       u16* __restrict__ ybf) {
  size_t idx = (size_t)blockIdx.x * 256 + threadIdx.x;
  if (idx >= (size_t)MROWS * DI) return;
  int d = (int)(idx & (DI - 1));
  size_t bl = idx >> 9;
  float z = xz[bl * (2 * DI) + DI + d];
  float v = (y[idx] + xs[idx] * Dsk[d]) * (z / (1.f + __expf(-z)));
  ybf[idx] = f2bf(v);
}

// ---------------- out = LN(x + r); optional bf16 copy ----------------
__global__ __launch_bounds__(256)
void k_add_ln(const float* __restrict__ x, const float* __restrict__ r,
              const float* __restrict__ g, const float* __restrict__ b,
              float* __restrict__ out, u16* __restrict__ outbf) {
  int row = blockIdx.x;
  int tid = threadIdx.x;
  int lane = tid & 31, wv = tid >> 5;
  const float* xr = x + (size_t)row * DM;
  const float* rr = r + (size_t)row * DM;
  float v0 = xr[tid] + rr[tid];
  float v1 = xr[tid + 256] + rr[tid + 256];
  __shared__ float red[8];
  float s = v0 + v1;
#pragma unroll
  for (int o = 16; o > 0; o >>= 1) s += __shfl_xor(s, o, 32);
  if (lane == 0) red[wv] = s;
  __syncthreads();
  float tot = 0.f;
#pragma unroll
  for (int i = 0; i < 8; i++) tot += red[i];
  float mu = tot * (1.f / DM);
  float d0 = v0 - mu, d1 = v1 - mu;
  float q = d0 * d0 + d1 * d1;
#pragma unroll
  for (int o = 16; o > 0; o >>= 1) q += __shfl_xor(q, o, 32);
  __syncthreads();
  if (lane == 0) red[wv] = q;
  __syncthreads();
  float totq = 0.f;
#pragma unroll
  for (int i = 0; i < 8; i++) totq += red[i];
  float inv = rsqrtf(totq * (1.f / DM) + 1e-5f);
  float o0 = d0 * inv * g[tid] + b[tid];
  float o1 = d1 * inv * g[tid + 256] + b[tid + 256];
  out[(size_t)row * DM + tid] = o0;
  out[(size_t)row * DM + tid + 256] = o1;
  if (outbf) {
    outbf[(size_t)row * DM + tid] = f2bf(o0);
    outbf[(size_t)row * DM + tid + 256] = f2bf(o1);
  }
}

// ---------------- orchestration ----------------
extern "C" void kernel_launch(void* const* d_in, const int* in_sizes, int n_in,
                              void* d_out, int out_size, void* d_ws, size_t ws_size,
                              hipStream_t stream) {
  const float* x_tokens  = (const float*)d_in[0];
  const float* in_proj_w = (const float*)d_in[1];
  const float* conv_w    = (const float*)d_in[2];
  const float* conv_b    = (const float*)d_in[3];
  const float* x_proj_w  = (const float*)d_in[4];
  const float* dt_proj_w = (const float*)d_in[5];
  const float* dt_proj_b = (const float*)d_in[6];
  const float* A_log     = (const float*)d_in[7];
  const float* D_skip    = (const float*)d_in[8];
  const float* out_proj_w= (const float*)d_in[9];
  const float* ln1_g     = (const float*)d_in[10];
  const float* ln1_b     = (const float*)d_in[11];
  const float* ffn_w1    = (const float*)d_in[12];
  const float* ffn_b1    = (const float*)d_in[13];
  const float* ffn_w2    = (const float*)d_in[14];
  const float* ffn_b2    = (const float*)d_in[15];
  const float* ln2_g     = (const float*)d_in[16];
  const float* ln2_b     = (const float*)d_in[17];
  float* out = (float*)d_out;

  char* p = (char*)d_ws;
  auto alloc = [&](size_t bytes) -> void* {
    void* r = (void*)p; p += (bytes + 255) & ~(size_t)255; return r;
  };
  const size_t M = (size_t)MROWS;

  u16*   xbf  = (u16*)  alloc(M * DM * 2);
  u16*   w_in = (u16*)  alloc((size_t)2 * DI * DM * 2);
  u16*   w_xp = (u16*)  alloc((size_t)XPN * DI * 2);
  u16*   w_dt = (u16*)  alloc((size_t)DI * DTR * 2);
  u16*   w_op = (u16*)  alloc((size_t)DM * DI * 2);
  u16*   w_f1 = (u16*)  alloc((size_t)DFF * DM * 2);
  u16*   w_f2 = (u16*)  alloc((size_t)DM * DFF * 2);
  float* xz   = (float*)alloc(M * 2 * DI * 4);
  float* xs   = (float*)alloc(M * DI * 4);
  u16*   xsbf = (u16*)  alloc(M * DI * 2);
  float* dbl  = (float*)alloc(M * XPN * 4);
  u16*   dtlbf= (u16*)  alloc(M * DTR * 2);
  float* dtb  = (float*)alloc(M * DI * 4);
  float* yb   = (float*)alloc(M * DI * 4);
  u16*   ybf  = (u16*)  alloc(M * DI * 2);
  float* attn = (float*)alloc(M * DM * 4);
  float* ln1  = (float*)alloc(M * DM * 4);
  u16*   ln1bf= (u16*)  alloc(M * DM * 2);
  float* f1   = (float*)alloc(M * DFF * 4);
  u16*   f1bf = (u16*)  alloc(M * DFF * 2);
  float* f2o  = (float*)alloc(M * DM * 4);

  auto cvt = [&](const float* s, u16* d, size_t n) {
    k_f32_to_bf16<<<dim3((unsigned)((n + 255) / 256)), dim3(256), 0, stream>>>(s, d, (int)n);
  };
  cvt(x_tokens,  xbf,  M * DM);
  cvt(in_proj_w, w_in, (size_t)2 * DI * DM);
  cvt(x_proj_w,  w_xp, (size_t)XPN * DI);
  cvt(dt_proj_w, w_dt, (size_t)DI * DTR);
  cvt(out_proj_w,w_op, (size_t)DM * DI);
  cvt(ffn_w1,    w_f1, (size_t)DFF * DM);
  cvt(ffn_w2,    w_f2, (size_t)DM * DFF);

  dim3 blk(256);
  auto gg = [&](int N) { return dim3((unsigned)((N + 63) / 64), (unsigned)((MROWS + 127) / 128)); };
  const unsigned nel = (unsigned)((M * DI + 255) / 256);

  // in_proj: xz = x @ W_in^T
  k_gemm_bf16<0><<<gg(2 * DI), blk, 0, stream>>>(xbf, w_in, nullptr, xz, MROWS, 2 * DI, DM);
  // depthwise conv + silu
  k_conv_silu<<<dim3(nel), dim3(256), 0, stream>>>(xz, conv_w, conv_b, xs, xsbf);
  // x_proj: dbl = xs @ W_xp^T
  k_gemm_bf16<0><<<gg(XPN), blk, 0, stream>>>(xsbf, w_xp, nullptr, dbl, MROWS, XPN, DI);
  // dtl (first 32 cols of dbl) -> bf16
  k_f32_to_bf16_strided<<<dim3((unsigned)((M * DTR + 255) / 256)), dim3(256), 0, stream>>>(
      dbl, dtlbf, MROWS, XPN, DTR);
  // dt = softplus(dtl @ W_dt^T + b)
  k_gemm_bf16<2><<<gg(DI), blk, 0, stream>>>(dtlbf, w_dt, dt_proj_b, dtb, MROWS, DI, DTR);
  // selective scan
  k_scan<<<dim3(BB * DI / 8), dim3(256), 0, stream>>>(dtb, dbl, xs, A_log, yb);
  // gate
  k_gate<<<dim3(nel), dim3(256), 0, stream>>>(yb, xs, xz, D_skip, ybf);
  // out_proj
  k_gemm_bf16<0><<<gg(DM), blk, 0, stream>>>(ybf, w_op, nullptr, attn, MROWS, DM, DI);
  // residual + LN1
  k_add_ln<<<dim3((unsigned)MROWS), dim3(256), 0, stream>>>(x_tokens, attn, ln1_g, ln1_b, ln1, ln1bf);
  // FFN
  k_gemm_bf16<1><<<gg(DFF), blk, 0, stream>>>(ln1bf, w_f1, ffn_b1, f1, MROWS, DFF, DM);
  cvt(f1, f1bf, M * DFF);
  k_gemm_bf16<0><<<gg(DM), blk, 0, stream>>>(f1bf, w_f2, ffn_b2, f2o, MROWS, DM, DFF);
  // residual + LN2 -> output
  k_add_ln<<<dim3((unsigned)MROWS), dim3(256), 0, stream>>>(ln1, f2o, ln2_g, ln2_b, out, nullptr);
}
